// MaxSimLoss_62139586838766
// MI455X (gfx1250) — matrix operand order repaired and checked by reading, not compiled, not run on hardware
//
#include <hip/hip_runtime.h>

typedef __attribute__((ext_vector_type(16))) _Float16 v16h;
typedef __attribute__((ext_vector_type(8)))  _Float16 v8h;
typedef __attribute__((ext_vector_type(4)))  _Float16 v4h;
typedef __attribute__((ext_vector_type(8)))  float    v8f;

#define NORM_EPS 1e-12f
#define MARGIN   0.1f

__device__ __forceinline__ float wave_sum32(float v) {
#pragma unroll
    for (int off = 16; off >= 1; off >>= 1) v += __shfl_xor(v, off, 32);
    return v;
}

// Load one 128-float row, L2-normalize (f32), store as f16 into LDS.
// All 32 lanes of a wave cooperate: lane handles 4 consecutive floats.
__device__ __forceinline__ void stage_row_f16(const float* __restrict__ src,
                                              _Float16* dst, int lane) {
    const float4 v = reinterpret_cast<const float4*>(src)[lane];
    float ss = v.x * v.x + v.y * v.y + v.z * v.z + v.w * v.w;
    ss = wave_sum32(ss);
    const float inv = 1.0f / fmaxf(sqrtf(ss), NORM_EPS);
    v4h h;
    h[0] = (_Float16)(v.x * inv);
    h[1] = (_Float16)(v.y * inv);
    h[2] = (_Float16)(v.z * inv);
    h[3] = (_Float16)(v.w * inv);
    *reinterpret_cast<v4h*>(dst + lane * 4) = h;
}

// One workgroup per batch. 256 threads = 8 waves.
// Waves: strip = w>>1 (anchor rows strip*16..+15), s-half = w&1.
__global__ __launch_bounds__(256)
void maxsim_batch_kernel(const float* __restrict__ anchor,
                         const float* __restrict__ pos,
                         const float* __restrict__ neg,
                         float* __restrict__ partial) {
    __shared__ __align__(16) _Float16 sA[64 * 128];    // 16 KB normalized anchor (f16)
    __shared__ __align__(16) _Float16 sB[256 * 128];   // 64 KB normalized pos/neg chunk (f16)
    __shared__ float sPart[8 * 16];                    // per-wave row-max partials
    __shared__ float sRow[64];
    __shared__ float sScore[2];

    const int b    = blockIdx.x;
    const int tid  = threadIdx.x;
    const int lane = tid & 31;
    const int w    = tid >> 5;
    const int ln15 = lane & 15;
    const int half = lane >> 4;

    // ---- Phase 1: normalize anchor into LDS (f16). Wave w: rows w*8..w*8+7.
    const float* aBase = anchor + (size_t)b * 64 * 128;
#pragma unroll
    for (int r = 0; r < 8; ++r) {
        const int row = w * 8 + r;
        stage_row_f16(aBase + row * 128, sA + row * 128, lane);
    }
    __syncthreads();

    // ---- Preload A fragments (constant per wave for whole kernel).
    // 16-bit A 16x32 layout: lane m = lane&15; koff = (lane>>4)*8;
    // halves 0-7 = K koff..koff+7, halves 8-15 = K koff+16..koff+23.
    const int strip = w >> 1;
    v16h aFrag[4];
#pragma unroll
    for (int kc = 0; kc < 4; ++kc) {
        const int base = (strip * 16 + ln15) * 128 + kc * 32 + half * 8;
        const v8h lo = *reinterpret_cast<const v8h*>(sA + base);
        const v8h hi = *reinterpret_cast<const v8h*>(sA + base + 16);
#pragma unroll
        for (int i = 0; i < 8; ++i) { aFrag[kc][i] = lo[i]; aFrag[kc][i + 8] = hi[i]; }
    }

#pragma unroll 1
    for (int which = 0; which < 2; ++which) {
        const float* mat = (which == 0 ? pos : neg) + (size_t)b * 512 * 128;

        v8f rmax;
#pragma unroll
        for (int r = 0; r < 8; ++r) rmax[r] = -3.0e38f;

#pragma unroll 1
        for (int chunk = 0; chunk < 2; ++chunk) {
            __syncthreads();   // previous sB contents fully consumed
            // ---- Stage + normalize 256 rows of pos/neg. Wave w: rows w*32..w*32+31.
            for (int r = 0; r < 32; ++r) {
                const int lrow = w * 32 + r;
                stage_row_f16(mat + (size_t)(chunk * 256 + lrow) * 128,
                              sB + lrow * 128, lane);
            }
            __syncthreads();

            // ---- WMMA: 8 s-tiles x 4 K-chunks per wave, fresh C per tile, running max.
#pragma unroll
            for (int t = 0; t < 8; ++t) {
                // dense B 32x16 layout: lane n = lane&15; koff = (lane>>4)*8; halves 0-7 = K koff..koff+7, 8-15 = K koff+16..koff+23
                const int lrow = ((w & 1) * 8 + t) * 16 + ln15;
                v8f c = {};
#pragma unroll
                for (int kc = 0; kc < 4; ++kc) {
                    const int base = lrow * 128 + kc * 32 + half * 8;
                    const v8h lo = *reinterpret_cast<const v8h*>(sB + base);
                    const v8h hi = *reinterpret_cast<const v8h*>(sB + base + 16);
                    v16h bFrag;
#pragma unroll
                    for (int i = 0; i < 8; ++i) { bFrag[i] = lo[i]; bFrag[i + 8] = hi[i]; }
                    c = __builtin_amdgcn_wmma_f32_16x16x32_f16(
                            false, aFrag[kc], false, bFrag,
                            (short)0, c, false, false);
                }
#pragma unroll
                for (int r = 0; r < 8; ++r) rmax[r] = fmaxf(rmax[r], c[r]);
            }
        }

        // ---- Row-max across the 16 lanes of each half (C layout: VGPR r -> rows M=r, M=r+8)
#pragma unroll
        for (int r = 0; r < 8; ++r) {
            float v = rmax[r];
            v = fmaxf(v, __shfl_xor(v, 8, 32));
            v = fmaxf(v, __shfl_xor(v, 4, 32));
            v = fmaxf(v, __shfl_xor(v, 2, 32));
            v = fmaxf(v, __shfl_xor(v, 1, 32));
            if (lane == 0)  sPart[w * 16 + r]     = v;  // row strip*16 + r
            if (lane == 16) sPart[w * 16 + 8 + r] = v;  // row strip*16 + 8 + r
        }
        __syncthreads();

        // Combine the two waves of each strip, then mean over 64 rows.
        if (tid < 64) {
            const int sbase = (tid >> 4) * 32 + (tid & 15);
            sRow[tid] = fmaxf(sPart[sbase], sPart[sbase + 16]);
        }
        __syncthreads();
        if (w == 0) {
            float s = sRow[lane] + sRow[lane + 32];
            s = wave_sum32(s);
            if (lane == 0) sScore[which] = s * (1.0f / 64.0f);
        }
        __syncthreads();
    }

    if (tid == 0)
        partial[b] = fmaxf(MARGIN + sScore[1] - sScore[0], 0.0f);
}

// Deterministic final reduction of 256 per-batch hinge terms.
__global__ __launch_bounds__(256)
void maxsim_reduce_kernel(const float* __restrict__ partial,
                          float* __restrict__ out) {
    __shared__ float sSum[8];
    const int tid = threadIdx.x, lane = tid & 31, w = tid >> 5;
    float v = partial[tid];
    v = wave_sum32(v);
    if (lane == 0) sSum[w] = v;
    __syncthreads();
    if (tid == 0) {
        float s = 0.0f;
#pragma unroll
        for (int i = 0; i < 8; ++i) s += sSum[i];
        out[0] = s * (1.0f / 256.0f);
    }
}

extern "C" void kernel_launch(void* const* d_in, const int* in_sizes, int n_in,
                              void* d_out, int out_size, void* d_ws, size_t ws_size,
                              hipStream_t stream) {
    const float* anchor = (const float*)d_in[0];
    const float* pos    = (const float*)d_in[1];
    const float* neg    = (const float*)d_in[2];
    float* out     = (float*)d_out;
    float* partial = (float*)d_ws;   // 256 floats of scratch

    maxsim_batch_kernel<<<256, 256, 0, stream>>>(anchor, pos, neg, partial);
    maxsim_reduce_kernel<<<1, 256, 0, stream>>>(partial, out);
}
